// Attentive_87746181857679
// MI455X (gfx1250) — matrix operand order repaired and checked by reading, not compiled, not run on hardware
//
#include <hip/hip_runtime.h>
#include <hip/hip_bf16.h>

// ---------------- problem constants (match reference) ----------------
constexpr int D  = 1024;          // latent
constexpr int H  = 512;           // hidden
constexpr int R  = 36;            // regions
constexpr int T  = 50;            // max caption length
constexpr int BI = 1024;          // images
constexpr int BC = 1024;          // captions
constexpr int M3 = BI * R;        // 36864 rows of the attention GEMM

typedef __attribute__((ext_vector_type(16))) _Float16 v16h;
typedef __attribute__((ext_vector_type(8)))  _Float16 v8h;
typedef __attribute__((ext_vector_type(4)))  _Float16 v4h;
typedef __attribute__((ext_vector_type(8)))  float    v8f;
typedef int v4i __attribute__((vector_size(16)));

#define AS1 __attribute__((address_space(1)))
#define AS3 __attribute__((address_space(3)))

// ---- CDNA5 async global->LDS staging (ASYNCcnt path), with fallbacks ----
__device__ __forceinline__ void async_g2l_b128(const _Float16* gsrc, _Float16* ldst) {
#if __has_builtin(__builtin_amdgcn_global_load_async_to_lds_b128)
  __builtin_amdgcn_global_load_async_to_lds_b128(
      (AS1 v4i*)(uintptr_t)gsrc, (AS3 v4i*)(uintptr_t)ldst, 0, 0);
#else
  asm volatile("global_load_async_to_lds_b128 %0, %1, off"
               :: "v"((unsigned)(uintptr_t)ldst), "v"(gsrc)
               : "memory");
#endif
}

__device__ __forceinline__ void wait_async_zero() {
#if __has_builtin(__builtin_amdgcn_s_wait_asynccnt)
  __builtin_amdgcn_s_wait_asynccnt(0);
#else
  asm volatile("s_wait_asynccnt 0" ::: "memory");
#endif
}

// Stage 16 contiguous rows (16*D halves = 32 KB) into LDS, 256 threads.
__device__ __forceinline__ void stage_16xD(const _Float16* __restrict__ src,
                                           _Float16* As) {
  const int tid = threadIdx.x;
#pragma unroll
  for (int it = 0; it < 8; ++it) {
    const int off = it * 2048 + tid * 8;          // 8 halves = 16 B per lane
    async_g2l_b128(src + off, As + off);
  }
  wait_async_zero();   // per-wave; __syncthreads() below makes it block-visible
  __syncthreads();
}

// Assemble a 16-half WMMA fragment from two 16-byte chunks.
__device__ __forceinline__ v16h mk16(const _Float16* p0, const _Float16* p1) {
  v8h lo = *(const v8h*)p0;
  v8h hi = *(const v8h*)p1;
  v16h r;
#pragma unroll
  for (int i = 0; i < 8; ++i) { r[i] = lo[i]; r[i + 8] = hi[i]; }
  return r;
}

// 256-thread block sum reduction (wave32).
__device__ __forceinline__ float blockReduceSum(float v, float* sred) {
#pragma unroll
  for (int m = 1; m < 32; m <<= 1) v += __shfl_xor(v, m, 32);
  int wid = threadIdx.x >> 5, ln = threadIdx.x & 31;
  if (ln == 0) sred[wid] = v;
  __syncthreads();
  if (wid == 0) {
    float t = (ln < 8) ? sred[ln] : 0.0f;
#pragma unroll
    for (int m = 1; m < 8; m <<= 1) t += __shfl_xor(t, m, 32);
    if (ln == 0) sred[0] = t;
  }
  __syncthreads();
  float r = sred[0];
  __syncthreads();
  return r;
}

__device__ __forceinline__ float leaky(float x) { return x >= 0.0f ? x : 0.1f * x; }

// ---------------- K1: captions -> txt_n (f16) and q = txt[0] ----------------
__global__ void __launch_bounds__(256)
k_txt(const float* __restrict__ cap, const int* __restrict__ lens,
      _Float16* __restrict__ txtn16, float* __restrict__ q) {
  __shared__ float sred[8];
  const int b = blockIdx.x, tid = threadIdx.x;
  const int len = lens[b];
  const float* capb = cap + (size_t)b * T * D;
  float a0 = 0.f, a1 = 0.f, a2 = 0.f, a3 = 0.f;
  for (int t = 0; t < len; ++t) {
    float4 x = *(const float4*)(capb + (size_t)t * D + tid * 4);
    float y0 = leaky(x.x), y1 = leaky(x.y), y2 = leaky(x.z), y3 = leaky(x.w);
    float ss = blockReduceSum(y0 * y0 + y1 * y1 + y2 * y2 + y3 * y3, sred);
    float sc = 1.0f / fmaxf(sqrtf(ss), 1e-12f);
    a0 += y0 * sc; a1 += y1 * sc; a2 += y2 * sc; a3 += y3 * sc;
  }
  const float inv = 1.0f / (float)len;
  a0 *= inv; a1 *= inv; a2 *= inv; a3 *= inv;
  if (b == 0) {
    float4 o = make_float4(a0, a1, a2, a3);
    *(float4*)(q + tid * 4) = o;
  }
  float ss2 = blockReduceSum(a0 * a0 + a1 * a1 + a2 * a2 + a3 * a3, sred);
  float sc2 = 1.0f / fmaxf(sqrtf(ss2), 1e-12f);
  _Float16* o = txtn16 + (size_t)b * D + tid * 4;
  o[0] = (_Float16)(a0 * sc2); o[1] = (_Float16)(a1 * sc2);
  o[2] = (_Float16)(a2 * sc2); o[3] = (_Float16)(a3 * sc2);
}

// ---------------- K2: image regions -> clipped_l2, stored f16 ----------------
__global__ void __launch_bounds__(256)
k_imgnorm(const float* __restrict__ img, _Float16* __restrict__ imgn16) {
  __shared__ float sred[8];
  const size_t row = blockIdx.x;
  const int tid = threadIdx.x;
  float4 x = *(const float4*)(img + row * D + tid * 4);
  float y0 = leaky(x.x), y1 = leaky(x.y), y2 = leaky(x.z), y3 = leaky(x.w);
  float ss = blockReduceSum(y0 * y0 + y1 * y1 + y2 * y2 + y3 * y3, sred);
  float sc = 1.0f / fmaxf(sqrtf(ss), 1e-12f);
  _Float16* o = imgn16 + row * D + tid * 4;
  o[0] = (_Float16)(y0 * sc); o[1] = (_Float16)(y1 * sc);
  o[2] = (_Float16)(y2 * sc); o[3] = (_Float16)(y3 * sc);
}

// ---------------- K0: w0 image-half -> f16 ----------------
__global__ void __launch_bounds__(256)
k_w0conv(const float* __restrict__ w0, _Float16* __restrict__ w0b16) {
  int idx = blockIdx.x * 256 + threadIdx.x;      // 0 .. H*D-1
  int h = idx >> 10, d = idx & (D - 1);
  w0b16[idx] = (_Float16)w0[(size_t)h * (2 * D) + D + d];
}

// ---------------- Kq: qb[h] = q . w0[h, :D] + b0[h] ----------------
__global__ void __launch_bounds__(256)
k_qdot(const float* __restrict__ q, const float* __restrict__ w0,
       const float* __restrict__ b0, float* __restrict__ qb) {
  __shared__ float sred[8];
  const int h = blockIdx.x, tid = threadIdx.x;
  float4 w = *(const float4*)(w0 + (size_t)h * (2 * D) + tid * 4);
  float4 qq = *(const float4*)(q + tid * 4);
  float p = w.x * qq.x + w.y * qq.y + w.z * qq.z + w.w * qq.w;
  p = blockReduceSum(p, sred);
  if (tid == 0) qb[h] = p + b0[h];
}

// ---------------- K3: big GEMM (36864x1024 @ 1024x512) + fused relu.w1 ----------------
// Block: 16-row M tile (A async-staged in LDS), 8 waves x 4 N-tiles = all 512 columns.
__global__ void __launch_bounds__(256)
k_attn(const _Float16* __restrict__ imgn16, const _Float16* __restrict__ w0b16,
       const float* __restrict__ qb, const float* __restrict__ w1,
       float* __restrict__ alpha_raw) {
  __shared__ __align__(16) _Float16 As[16 * D];   // 32 KB
  __shared__ float aAcc[16];
  const int tid = threadIdx.x;
  const size_t m0 = (size_t)blockIdx.x * 16;

  if (tid < 16) aAcc[tid] = 0.0f;
  stage_16xD(imgn16 + m0 * D, As);

  const int wave = tid >> 5, lane = tid & 31, r = lane & 15;
  const bool hiHalf = lane >= 16;

  // one base pointer per wave; the 4 N-tile streams live at constant offsets
  // (j*16*D halves) so the backend can use base+immediate addressing.
  const _Float16* bbase =
      w0b16 + (size_t)((wave << 6) + r) * D + (hiHalf ? 8 : 0);
  const _Float16* abase = As + r * D + (hiHalf ? 8 : 0);

  v8f c[4] = {};
  for (int k0 = 0; k0 < D; k0 += 32) {
    v16h a = mk16(abase + k0, abase + k0 + 16);
#pragma unroll
    for (int j = 0; j < 4; ++j) {
      const _Float16* bp = bbase + k0 + j * (16 * D);
      v16h b = mk16(bp, bp + 16);
      c[j] = __builtin_amdgcn_wmma_f32_16x16x32_f16(
          false, a, false, b, (short)0, c[j], false, false);
    }
  }

  // epilogue: alpha_m += sum_n relu(c + qb[n]) * w1[n]
  float p[8] = {0.f, 0.f, 0.f, 0.f, 0.f, 0.f, 0.f, 0.f};
#pragma unroll
  for (int j = 0; j < 4; ++j) {
    const int col = ((wave << 2) + j) * 16 + r;
    const float qv = qb[col], wv = w1[col];
#pragma unroll
    for (int rr = 0; rr < 8; ++rr)
      p[rr] += fmaxf(c[j][rr] + qv, 0.0f) * wv;
  }
#pragma unroll
  for (int rr = 0; rr < 8; ++rr)
#pragma unroll
    for (int m = 1; m < 16; m <<= 1) p[rr] += __shfl_xor(p[rr], m, 16);
  if (r == 0) {   // lane 0 holds rows 0..7, lane 16 holds rows 8..15
    const int mb = hiHalf ? 8 : 0;
#pragma unroll
    for (int rr = 0; rr < 8; ++rr) atomicAdd(&aAcc[mb + rr], p[rr]);
  }
  __syncthreads();
  if (tid < 16) alpha_raw[m0 + tid] = aAcc[tid];   // b1 dropped: softmax-invariant
}

// ---------------- K4: softmax over regions + weighted sum + l2norm ----------------
__global__ void __launch_bounds__(256)
k_pool(const float* __restrict__ alpha_raw, const _Float16* __restrict__ imgn16,
       _Float16* __restrict__ imgOutN16) {
  __shared__ float al[R];
  __shared__ float sred[8];
  const int i = blockIdx.x, tid = threadIdx.x;
  if (tid == 0) {
    float mx = -1e30f;
    for (int rr = 0; rr < R; ++rr) { al[rr] = alpha_raw[i * R + rr]; mx = fmaxf(mx, al[rr]); }
    float s = 0.0f;
    for (int rr = 0; rr < R; ++rr) { al[rr] = __expf(al[rr] - mx); s += al[rr]; }
    const float inv = 1.0f / s;
    for (int rr = 0; rr < R; ++rr) al[rr] *= inv;
  }
  __syncthreads();
  float a0 = 0.f, a1 = 0.f, a2 = 0.f, a3 = 0.f;
  const _Float16* base = imgn16 + (size_t)i * R * D + tid * 4;
  for (int rr = 0; rr < R; ++rr) {
    v4h x = *(const v4h*)(base + rr * D);
    const float w = al[rr];
    a0 += w * (float)x[0]; a1 += w * (float)x[1];
    a2 += w * (float)x[2]; a3 += w * (float)x[3];
  }
  float ss = blockReduceSum(a0 * a0 + a1 * a1 + a2 * a2 + a3 * a3, sred);
  float sc = 1.0f / fmaxf(sqrtf(ss), 1e-12f);
  _Float16* o = imgOutN16 + (size_t)i * D + tid * 4;
  o[0] = (_Float16)(a0 * sc); o[1] = (_Float16)(a1 * sc);
  o[2] = (_Float16)(a2 * sc); o[3] = (_Float16)(a3 * sc);
}

// ---------------- K5: sims = img_out_n @ txt_n^T  (1024x1024x1024) ----------------
__global__ void __launch_bounds__(256)
k_sims(const _Float16* __restrict__ A16, const _Float16* __restrict__ B16,
       float* __restrict__ out) {
  __shared__ __align__(16) _Float16 As[16 * D];   // 32 KB
  const int tid = threadIdx.x;
  const size_t m0 = (size_t)blockIdx.x * 16;
  stage_16xD(A16 + m0 * D, As);

  const int wave = tid >> 5, lane = tid & 31, r = lane & 15;
  const bool hiHalf = lane >= 16;
  const int n0 = blockIdx.y * 128 + wave * 16;

  const _Float16* bbase = B16 + (size_t)(n0 + r) * D + (hiHalf ? 8 : 0);
  const _Float16* abase = As + r * D + (hiHalf ? 8 : 0);

  v8f c = {};
  for (int k0 = 0; k0 < D; k0 += 32) {
    v16h a = mk16(abase + k0, abase + k0 + 16);
    v16h b = mk16(bbase + k0, bbase + k0 + 16);
    c = __builtin_amdgcn_wmma_f32_16x16x32_f16(false, a, false, b, (short)0, c, false, false);
  }
  const int n = n0 + r;
  const size_t mb = m0 + (hiHalf ? 8 : 0);
#pragma unroll
  for (int rr = 0; rr < 8; ++rr)
    out[(mb + rr) * BC + n] = c[rr];
}

// ---------------- host launch ----------------
extern "C" void kernel_launch(void* const* d_in, const int* in_sizes, int n_in,
                              void* d_out, int out_size, void* d_ws, size_t ws_size,
                              hipStream_t stream) {
  const float* img  = (const float*)d_in[0];   // (BI, R, D)
  const float* cap  = (const float*)d_in[1];   // (BC, T, D)
  const int*   lens = (const int*)d_in[2];     // (BC,)
  const float* w0   = (const float*)d_in[3];   // (H, 2D)
  const float* b0   = (const float*)d_in[4];   // (H,)
  const float* w1   = (const float*)d_in[5];   // (1, H)
  // b1 (d_in[6]) is a uniform shift before softmax -> no effect on output.
  float* out = (float*)d_out;

  char* p = (char*)d_ws;
  auto alloc = [&](size_t bytes) {
    char* r = p;
    p += (bytes + 255) & ~(size_t)255;
    return r;
  };
  _Float16* txtn16    = (_Float16*)alloc((size_t)BC * D * 2);
  float*    q         = (float*)   alloc((size_t)D * 4);
  _Float16* w0b16     = (_Float16*)alloc((size_t)H * D * 2);
  float*    qb        = (float*)   alloc((size_t)H * 4);
  _Float16* imgn16    = (_Float16*)alloc((size_t)M3 * D * 2);
  float*    alpha_raw = (float*)   alloc((size_t)M3 * 4);
  _Float16* imgOutN16 = (_Float16*)alloc((size_t)BI * D * 2);

  k_txt    <<<BC,               256, 0, stream>>>(cap, lens, txtn16, q);
  k_w0conv <<<(H * D) / 256,    256, 0, stream>>>(w0, w0b16);
  k_imgnorm<<<M3,               256, 0, stream>>>(img, imgn16);
  k_qdot   <<<H,                256, 0, stream>>>(q, w0, b0, qb);
  k_attn   <<<M3 / 16,          256, 0, stream>>>(imgn16, w0b16, qb, w1, alpha_raw);
  k_pool   <<<BI,               256, 0, stream>>>(alpha_raw, imgn16, imgOutN16);
  k_sims   <<<dim3(BI / 16, BC / 128), 256, 0, stream>>>(imgOutN16, txtn16, out);
}
